// CausalSelfAttention_47562467836684
// MI455X (gfx1250) — matrix operand order repaired and checked by reading, not compiled, not run on hardware
//
#include <hip/hip_runtime.h>
#include <cmath>

typedef __attribute__((ext_vector_type(16))) __bf16 v16bf;
typedef __attribute__((ext_vector_type(8)))  __bf16 v8bf;
typedef __attribute__((ext_vector_type(8)))  float  v8f;

#define NTOK 4096   // B*T
#define BB   2
#define TT   2048
#define CC   1024
#define HH   16
#define DD   64
#define HDP  8      // head pairs
#define GCN  32
#define WIN  1024
#define NEGF -3.0e38f
#define RMSEPS 1.1920929e-07f
#define QSCALE 0.125f  // 1/sqrt(64), folded into Q at prep time

// ---------------------------------------------------------------------------
// Fragment helpers (wave32; A/B use contiguous k = 16h+s, not the ISA table order)
// A 16-bit 16x32: lane m=l&15, half h=l>>4; elems 0..7 = K[16h..16h+7],
//                 elems 8..15 = K[16h+8..16h+15]  (two contiguous 16B runs)
// B 16-bit 32x16: lane n=l&15, half h; elems 0..15 = K[16h..16h+15] (contig 32B)
// C/D f32 16x16:  lane n=l&15; vgpr r -> row m = r + 8*(l>>4)
// ---------------------------------------------------------------------------
__device__ __forceinline__ v16bf mk_frag(v8bf lo, v8bf hi) {
  v16bf r;
#pragma unroll
  for (int i = 0; i < 8; i++) { r[i] = lo[i]; r[8 + i] = hi[i]; }
  return r;
}
// ptr pre-offset by +16h elements; runs at +0 and +8
__device__ __forceinline__ v16bf load_afrag(const __bf16* p) {
  return mk_frag(*(const v8bf*)(p), *(const v8bf*)(p + 8));
}
// ptr pre-offset by +16h elements; runs at +0 and +8
__device__ __forceinline__ v16bf load_bfrag(const __bf16* p) {
  return mk_frag(*(const v8bf*)(p), *(const v8bf*)(p + 8));
}

__device__ __forceinline__ v8f wmma_bf16(v16bf a, v16bf b, v8f c) {
  return __builtin_amdgcn_wmma_f32_16x16x32_bf16(false, a, false, b, (short)0, c,
                                                 false, false);
}

__device__ __forceinline__ float rowred_max(float v) {
  v = fmaxf(v, __shfl_xor(v, 1));
  v = fmaxf(v, __shfl_xor(v, 2));
  v = fmaxf(v, __shfl_xor(v, 4));
  v = fmaxf(v, __shfl_xor(v, 8));
  return v;
}
__device__ __forceinline__ float rowred_sum(float v) {
  v += __shfl_xor(v, 1);
  v += __shfl_xor(v, 2);
  v += __shfl_xor(v, 4);
  v += __shfl_xor(v, 8);
  return v;
}

// ---------------------------------------------------------------------------
// Conversion kernels
// ---------------------------------------------------------------------------
__global__ void f32_to_bf16(const float* __restrict__ in, __bf16* __restrict__ out,
                            int n) {
  int i = blockIdx.x * blockDim.x + threadIdx.x;
  if (i < n) out[i] = (__bf16)in[i];
}

// in [rows][cols] -> out [cols][rows]  (rows == cols == CC here)
__global__ void f32_to_bf16_T(const float* __restrict__ in,
                              __bf16* __restrict__ out) {
  int i = blockIdx.x * blockDim.x + threadIdx.x;
  int r = i >> 10, c = i & (CC - 1);
  out[(size_t)c * CC + r] = (__bf16)in[i];
}

// ---------------------------------------------------------------------------
// bf16 WMMA GEMM, fixed 4096x1024x1024: C = A[M,K] * Bt[N,K]^T.
// 256 thr = 8 waves, block tile 128x128, wave tile 32x64, K-step 32.
// All fragment addresses are base-pointer + compile-time immediate offsets.
// ---------------------------------------------------------------------------
__global__ __launch_bounds__(256) void gemm_bf16_wmma(
    const __bf16* __restrict__ A, const __bf16* __restrict__ Bt,
    float* __restrict__ C) {
  constexpr int N = CC, K = CC;
  const int lane = threadIdx.x & 31;
  const int wave = threadIdx.x >> 5;
  const int wm = wave >> 1, wn = wave & 1;
  const int h = lane >> 4, ln = lane & 15;
  const int m0 = blockIdx.x * 128 + wm * 32;
  const int n0 = blockIdx.y * 128 + wn * 64;

  v8f acc[2][4];
#pragma unroll
  for (int im = 0; im < 2; im++)
#pragma unroll
    for (int in_ = 0; in_ < 4; in_++)
#pragma unroll
      for (int r = 0; r < 8; r++) acc[im][in_][r] = 0.f;

  const __bf16* ap = A + (size_t)(m0 + ln) * K + 16 * h;
  const __bf16* bp = Bt + (size_t)(n0 + ln) * K + 16 * h;

  for (int kb = 0; kb < K; kb += 32) {
    v16bf a[2], b[4];
#pragma unroll
    for (int im = 0; im < 2; im++)
      a[im] = load_afrag(ap + im * 16 * K);
#pragma unroll
    for (int in_ = 0; in_ < 4; in_++)
      b[in_] = load_bfrag(bp + in_ * 16 * K);
#pragma unroll
    for (int im = 0; im < 2; im++)
#pragma unroll
      for (int in_ = 0; in_ < 4; in_++)
        acc[im][in_] = wmma_bf16(a[im], b[in_], acc[im][in_]);
    ap += 32;
    bp += 32;
  }

  float* cp = C + (size_t)(m0 + 8 * h) * N + n0 + ln;
#pragma unroll
  for (int im = 0; im < 2; im++)
#pragma unroll
    for (int in_ = 0; in_ < 4; in_++)
#pragma unroll
      for (int r = 0; r < 8; r++)
        cp[(16 * im + r) * N + 16 * in_] = acc[im][in_][r];
}

// ---------------------------------------------------------------------------
// Fused gate + RoPE + RMS prep.
// grid (B*H, T/256), block 256, one thread per (b,h,t).
// qr/kr: [b][h][t][d] bf16 (Q pre-scaled by 1/sqrt(D));
// vt:    [b][h][d][t] bf16 (transposed so PV B-frags are contiguous)
// ---------------------------------------------------------------------------
__device__ __forceinline__ void rope_rms_store(const float* __restrict__ src,
                                               const float* __restrict__ cp,
                                               const float* __restrict__ sp,
                                               __bf16* __restrict__ dst,
                                               float outscale) {
  float a[DD];
#pragma unroll
  for (int d = 0; d < 32; d++) {
    float x1 = src[d], x2 = src[32 + d];
    float c = cp[d], s = sp[d];
    a[d] = x1 * c + x2 * s;
    a[32 + d] = -x1 * s + x2 * c;
  }
  float ss = 0.f;
#pragma unroll
  for (int d = 0; d < DD; d++) ss += a[d] * a[d];
  float inv = rsqrtf(ss * (1.f / DD) + RMSEPS) * outscale;
#pragma unroll
  for (int d = 0; d < DD; d++) dst[d] = (__bf16)(a[d] * inv);
}

__global__ __launch_bounds__(256) void qkv_prep(
    const float* __restrict__ qf, const float* __restrict__ kf,
    const float* __restrict__ vf, const float* __restrict__ x,
    const float* __restrict__ ve, const float* __restrict__ cosA,
    const float* __restrict__ sinA, const float* __restrict__ Wg,
    __bf16* __restrict__ qr, __bf16* __restrict__ kr, __bf16* __restrict__ vt) {
  const int b = blockIdx.x / HH;
  const int hh = blockIdx.x % HH;
  const int t = blockIdx.y * 256 + threadIdx.x;
  const size_t bt = (size_t)b * TT + t;
  const float* cp = cosA + (size_t)t * 32;
  const float* sp = sinA + (size_t)t * 32;
  const size_t bh = (size_t)b * HH + hh;

  rope_rms_store(qf + bt * CC + hh * DD, cp, sp, qr + (bh * TT + t) * DD, QSCALE);
  rope_rms_store(kf + bt * CC + hh * DD, cp, sp, kr + (bh * TT + t) * DD, 1.f);

  float g = 0.f;
  const float* xr = x + bt * CC;
#pragma unroll
  for (int gi = 0; gi < GCN; gi++) g += xr[gi] * Wg[gi * HH + hh];
  g = 2.f / (1.f + __expf(-g));

  const float* vrow = vf + bt * CC + hh * DD;
  const float* verow = ve + bt * CC + hh * DD;
  __bf16* vtp = vt + bh * DD * TT + t;
#pragma unroll
  for (int d = 0; d < DD; d++)
    vtp[(size_t)d * TT] = (__bf16)(vrow[d] + g * verow[d]);
}

// ---------------------------------------------------------------------------
// lambda scalar
// ---------------------------------------------------------------------------
__global__ void lambda_kernel(const float* __restrict__ lq1,
                              const float* __restrict__ lk1,
                              const float* __restrict__ lq2,
                              const float* __restrict__ lk2, float lambda_init,
                              float* __restrict__ lam_out) {
  float d1 = 0.f, d2 = 0.f;
  for (int i = 0; i < DD; i++) { d1 += lq1[i] * lk1[i]; d2 += lq2[i] * lk2[i]; }
  d1 = fminf(fmaxf(d1, -10.f), 10.f);
  d2 = fminf(fmaxf(d2, -10.f), 10.f);
  *lam_out = __expf(d1) - __expf(d2) + lambda_init;
}

// ---------------------------------------------------------------------------
// Differential flash attention (windowed causal).
// grid (B*HDP, T/64), block 128 = 4 waves; wave = 16 rows x 128 out cols.
// Per 32-col step: 8 score WMMAs (q1k1,q2k2), 16 PV WMMAs (shared V frags).
// P transpose via wave-private LDS (16x40 bf16), in-order DS, no barriers.
// ---------------------------------------------------------------------------
__device__ __forceinline__ void online_update(v8f s[2], float* mrun, float* lrun,
                                              v8f* acc, __bf16 (*prow)[40], int h,
                                              int ln) {
#pragma unroll
  for (int r = 0; r < 8; r++) {
    float mx = rowred_max(fmaxf(s[0][r], s[1][r]));
    float nm = fmaxf(mrun[r], mx);
    float alpha = __expf(mrun[r] - nm);
    float p0 = __expf(s[0][r] - nm);
    float p1 = __expf(s[1][r] - nm);
    float rs = rowred_sum(p0 + p1);
    lrun[r] = lrun[r] * alpha + rs;
    mrun[r] = nm;
    prow[r + 8 * h][ln] = (__bf16)p0;
    prow[r + 8 * h][16 + ln] = (__bf16)p1;
#pragma unroll
    for (int nt = 0; nt < 8; nt++) acc[nt][r] *= alpha;
  }
}

__global__ __launch_bounds__(128) void diff_attn(
    const __bf16* __restrict__ qr, const __bf16* __restrict__ kr,
    const __bf16* __restrict__ vt, const float* __restrict__ dnw,
    const float* __restrict__ lam_p, __bf16* __restrict__ yn,
    float lambda_init) {
  __shared__ __bf16 pbuf[4][16][40];
  const int lane = threadIdx.x & 31;
  const int wave = threadIdx.x >> 5;
  const int h = lane >> 4, ln = lane & 15;
  const int b = blockIdx.x >> 3;
  const int hp = blockIdx.x & 7;
  const int t0 = blockIdx.y * 64;
  const int mbase = t0 + wave * 16;
  const int head1 = hp * 2;

  // head2 data sits at a fixed +TT*DD element offset from head1 (immediate).
  constexpr int H2 = TT * DD;
  const __bf16* q1p = qr + ((size_t)(b * HH + head1) * TT) * DD;
  const __bf16* k1p = kr + ((size_t)(b * HH + head1) * TT) * DD;
  const __bf16* v1p = vt + (size_t)(b * HH + head1) * DD * TT;

  const int mrow = mbase + ln;
  v16bf qa1[2], qa2[2];
  {
    const __bf16* qp = q1p + (size_t)mrow * DD + 16 * h;
    qa1[0] = load_afrag(qp);
    qa1[1] = load_afrag(qp + 32);
    qa2[0] = load_afrag(qp + H2);
    qa2[1] = load_afrag(qp + H2 + 32);
  }

  v8f acc1[8], acc2[8];
  float m1[8], l1[8], m2[8], l2[8];
#pragma unroll
  for (int nt = 0; nt < 8; nt++)
#pragma unroll
    for (int r = 0; r < 8; r++) { acc1[nt][r] = 0.f; acc2[nt][r] = 0.f; }
#pragma unroll
  for (int r = 0; r < 8; r++) { m1[r] = NEGF; l1[r] = 0.f; m2[r] = NEGF; l2[r] = 0.f; }

  int lo = mbase - WIN;
  if (lo < 0) lo = 0;
  lo &= ~31;

  // per-lane invariant bases; all per-nt/ch selection is immediate offsets
  const __bf16* kbase = k1p + (size_t)ln * DD + 16 * h;  // + j*DD per col
  const __bf16* vbase0 = v1p + (size_t)((ln << 1 | h) << 4);  // (16*ln+16h)? no:
  // careful: V B-frag column index n=ln along d, K(t) run starts at 16h:
  const __bf16* vbase = v1p + (size_t)ln * TT + 16 * h;

  for (int jb = lo; jb <= mbase + 15; jb += 32) {
    v8f s1[2], s2[2];
    const __bf16* kj = kbase + (size_t)jb * DD;
#pragma unroll
    for (int ch = 0; ch < 2; ch++) {
      const __bf16* kc = kj + ch * 16 * DD;
      v16bf kb10 = load_bfrag(kc);
      v16bf kb11 = load_bfrag(kc + 32);
      v16bf kb20 = load_bfrag(kc + H2);
      v16bf kb21 = load_bfrag(kc + H2 + 32);
      v8f s;
#pragma unroll
      for (int r = 0; r < 8; r++) s[r] = 0.f;
      s = wmma_bf16(qa1[0], kb10, s);
      s = wmma_bf16(qa1[1], kb11, s);
      s1[ch] = s;
#pragma unroll
      for (int r = 0; r < 8; r++) s[r] = 0.f;
      s = wmma_bf16(qa2[0], kb20, s);
      s = wmma_bf16(qa2[1], kb21, s);
      s2[ch] = s;
    }
    // mask (Q pre-scaled, so only select/neg-inf here)
#pragma unroll
    for (int ch = 0; ch < 2; ch++)
#pragma unroll
      for (int r = 0; r < 8; r++) {
        const int i = mbase + r + 8 * h;
        const int j = jb + 16 * ch + ln;
        const bool ok = (unsigned)(i - j) <= (unsigned)WIN;
        s1[ch][r] = ok ? s1[ch][r] : NEGF;
        s2[ch][r] = ok ? s2[ch][r] : NEGF;
      }

    online_update(s1, m1, l1, acc1, pbuf[wave], h, ln);
    v16bf pa1 = load_afrag(&pbuf[wave][ln][16 * h]);
    online_update(s2, m2, l2, acc2, pbuf[wave], h, ln);
    v16bf pa2 = load_afrag(&pbuf[wave][ln][16 * h]);

    const __bf16* vj = vbase + jb;
#pragma unroll
    for (int nt = 0; nt < 8; nt++) {
      // nt<4 -> v1 tiles, nt>=4 -> v2 tiles; all immediate offsets off vj
      v16bf vb = load_bfrag(vj + (nt & 3) * 16 * TT + (nt >> 2) * H2);
      acc1[nt] = wmma_bf16(pa1, vb, acc1[nt]);
      acc2[nt] = wmma_bf16(pa2, vb, acc2[nt]);
    }
  }
  (void)vbase0;

  const float lam = *lam_p;
  const float oscale = 1.f - lambda_init;
  float w[8];
#pragma unroll
  for (int nt = 0; nt < 8; nt++) w[nt] = dnw[nt * 16 + ln];

  __bf16* yrow = yn + ((size_t)b * TT + mbase + 8 * h) * CC + hp * 128 + ln;
#pragma unroll
  for (int r = 0; r < 8; r++) {
    float i1 = 1.f / l1[r], i2 = 1.f / l2[r];
    float sq = 0.f;
#pragma unroll
    for (int nt = 0; nt < 8; nt++) {
      float y = acc1[nt][r] * i1 - lam * acc2[nt][r] * i2;
      acc1[nt][r] = y;
      sq += y * y;
    }
    sq = rowred_sum(sq);
    float rn = rsqrtf(sq * (1.f / 128.f) + RMSEPS);
#pragma unroll
    for (int nt = 0; nt < 8; nt++)
      yrow[r * CC + nt * 16] = (__bf16)(acc1[nt][r] * rn * w[nt] * oscale);
  }
}

// ---------------------------------------------------------------------------
// Host launcher
// ---------------------------------------------------------------------------
extern "C" void kernel_launch(void* const* d_in, const int* in_sizes, int n_in,
                              void* d_out, int out_size, void* d_ws, size_t ws_size,
                              hipStream_t stream) {
  (void)in_sizes; (void)n_in; (void)out_size; (void)ws_size;

  const float* x = (const float*)d_in[0];
  const float* ve = (const float*)d_in[1];
  const float* cosA = (const float*)d_in[2];
  const float* sinA = (const float*)d_in[3];
  const float* Wq = (const float*)d_in[4];
  const float* Wk = (const float*)d_in[5];
  const float* Wv = (const float*)d_in[6];
  const float* Wp = (const float*)d_in[7];
  const float* Wg = (const float*)d_in[8];
  const float* lq1 = (const float*)d_in[9];
  const float* lk1 = (const float*)d_in[10];
  const float* lq2 = (const float*)d_in[11];
  const float* lk2 = (const float*)d_in[12];
  const float* dnw = (const float*)d_in[13];
  float* out = (float*)d_out;

  char* ws = (char*)d_ws;
  size_t off = 0;
  auto alloc = [&](size_t bytes) -> void* {
    void* p = ws + off;
    off += (bytes + 255) & ~(size_t)255;
    return p;
  };
  __bf16* xb = (__bf16*)alloc((size_t)NTOK * CC * 2);
  __bf16* wqT = (__bf16*)alloc((size_t)CC * CC * 2);
  __bf16* wkT = (__bf16*)alloc((size_t)CC * CC * 2);
  __bf16* wvT = (__bf16*)alloc((size_t)CC * CC * 2);
  __bf16* wpT = (__bf16*)alloc((size_t)CC * CC * 2);
  float* qf = (float*)alloc((size_t)NTOK * CC * 4);
  float* kf = (float*)alloc((size_t)NTOK * CC * 4);
  float* vf = (float*)alloc((size_t)NTOK * CC * 4);
  __bf16* qr = (__bf16*)alloc((size_t)BB * HH * TT * DD * 2);
  __bf16* kr = (__bf16*)alloc((size_t)BB * HH * TT * DD * 2);
  __bf16* vt = (__bf16*)alloc((size_t)BB * HH * DD * TT * 2);
  __bf16* yn = (__bf16*)alloc((size_t)NTOK * CC * 2);
  float* lam = (float*)alloc(16);

  const float LAMBDA_INIT = 0.8f - 0.6f * expf(-0.3f * 11.f);

  const int nx = NTOK * CC;
  f32_to_bf16<<<(nx + 255) / 256, 256, 0, stream>>>(x, xb, nx);
  const int nw = CC * CC;
  f32_to_bf16_T<<<nw / 256, 256, 0, stream>>>(Wq, wqT);
  f32_to_bf16_T<<<nw / 256, 256, 0, stream>>>(Wk, wkT);
  f32_to_bf16_T<<<nw / 256, 256, 0, stream>>>(Wv, wvT);
  f32_to_bf16_T<<<nw / 256, 256, 0, stream>>>(Wp, wpT);

  dim3 gg(NTOK / 128, CC / 128);
  gemm_bf16_wmma<<<gg, 256, 0, stream>>>(xb, wqT, qf);
  gemm_bf16_wmma<<<gg, 256, 0, stream>>>(xb, wkT, kf);
  gemm_bf16_wmma<<<gg, 256, 0, stream>>>(xb, wvT, vf);

  qkv_prep<<<dim3(BB * HH, TT / 256), 256, 0, stream>>>(qf, kf, vf, x, ve, cosA,
                                                        sinA, Wg, qr, kr, vt);
  lambda_kernel<<<1, 1, 0, stream>>>(lq1, lk1, lq2, lk2, LAMBDA_INIT, lam);

  diff_attn<<<dim3(BB * HDP, TT / 64), 128, 0, stream>>>(qr, kr, vt, dnw, lam, yn,
                                                         LAMBDA_INIT);

  gemm_bf16_wmma<<<gg, 256, 0, stream>>>(yn, wpT, out);
}
